// SVDQuantLinear_68942815036086
// MI455X (gfx1250) — matrix operand order repaired and checked by reading, not compiled, not run on hardware
//
#include <hip/hip_runtime.h>

// MI455X (gfx1250) AdaLoRA 4-bit quantized linear: out = x @ dequant(W)^T + t @ B^T.
// v4: all dtype conversion hoisted to one-shot bandwidth-bound pre-passes
// (x->bf16, W dequant->bf16, lora_B->bf16). Hot GEMM stages tiles with CDNA5
// async DMA (global_load_async_to_lds_b128, ASYNCcnt) and runs pure
// ds_load_b128 + v_wmma_f32_16x16x32_bf16. bf16 conversion now uses native
// (__bf16) casts hoping for hardware v_cvt selection. Fused fallback for small ws.

#define IN_F   4096
#define OUT_F  4096
#define GROUPQ 128
#define NGRP   (IN_F / GROUPQ)
#define RANK   16
#define LDSW   72   // LDS row stride in ushorts: 144B, 16B-aligned, spreads banks

typedef __attribute__((ext_vector_type(16))) __bf16 v16bf;
typedef __attribute__((ext_vector_type(2)))  __bf16 v2bf;
typedef __attribute__((ext_vector_type(8)))  float  v8f;
typedef __attribute__((address_space(3)))    unsigned char lds_byte_t;

union Frag { v16bf v; uint4 u[2]; };

__device__ __forceinline__ unsigned int pk2(float a, float b) {
  v2bf t;
  t[0] = (__bf16)a;            // fptrunc f32->bf16; native cvt if gfx1250 has it
  t[1] = (__bf16)b;
  return __builtin_bit_cast(unsigned int, t);
}
__device__ __forceinline__ uint4 pack8(const float* p) {
  float4 f0 = *(const float4*)p;
  float4 f1 = *(const float4*)(p + 4);
  uint4 o;
  o.x = pk2(f0.x, f0.y); o.y = pk2(f0.z, f0.w);
  o.z = pk2(f1.x, f1.y); o.w = pk2(f1.z, f1.w);
  return o;
}
__device__ __forceinline__ uint4 pack8s(const float* p, float s) {
  float4 f0 = *(const float4*)p;
  float4 f1 = *(const float4*)(p + 4);
  uint4 o;
  o.x = pk2(f0.x * s, f0.y * s); o.y = pk2(f0.z * s, f0.w * s);
  o.z = pk2(f1.x * s, f1.y * s); o.w = pk2(f1.z * s, f1.w * s);
  return o;
}
__device__ __forceinline__ uint4 dq8(const int* q, float s, float zs) {
  int4 q0 = *(const int4*)q;
  int4 q1 = *(const int4*)(q + 4);
  uint4 o;
  o.x = pk2(fmaf((float)q0.x, s, zs), fmaf((float)q0.y, s, zs));
  o.y = pk2(fmaf((float)q0.z, s, zs), fmaf((float)q0.w, s, zs));
  o.z = pk2(fmaf((float)q1.x, s, zs), fmaf((float)q1.y, s, zs));
  o.w = pk2(fmaf((float)q1.z, s, zs), fmaf((float)q1.w, s, zs));
  return o;
}

// CDNA5 async DMA: copy 16 bytes global -> LDS, tracked by ASYNCcnt.
// GVS mode: LDS dest address VGPR, 32-bit voffset VGPR, uniform SGPR64 base.
__device__ __forceinline__ void async_cp16(unsigned short* lds_dst,
                                           const unsigned short* gbase,
                                           unsigned voff) {
  lds_byte_t* lp = (lds_byte_t*)(void*)lds_dst;
  asm volatile("global_load_async_to_lds_b128 %0, %1, %2 offset:0"
               :: "v"(lp), "v"(voff), "s"(gbase)
               : "memory");
}
__device__ __forceinline__ void wait_async0() {
#if __has_builtin(__builtin_amdgcn_s_wait_asynccnt)
  __builtin_amdgcn_s_wait_asynccnt(0);
#else
  asm volatile("s_wait_asynccnt 0" ::: "memory");
#endif
}

// ---------------------------------------------------------------------------
// Pre-pass A: x fp32 -> bf16 (each element converted exactly once)
// ---------------------------------------------------------------------------
__global__ __launch_bounds__(256) void convert_x_kernel(
    const float* __restrict__ x, unsigned short* __restrict__ xbf) {
  size_t i = ((size_t)blockIdx.x * 256 + threadIdx.x) * 16;
  uint4* o = (uint4*)(xbf + i);
  o[0] = pack8(x + i);
  o[1] = pack8(x + i + 8);
}

// ---------------------------------------------------------------------------
// Pre-pass B: int4 codes -> bf16 dequant (each element dequantized once)
// ---------------------------------------------------------------------------
__global__ __launch_bounds__(256) void dequant_w_kernel(
    const int* __restrict__ qweight, const float* __restrict__ scales,
    const float* __restrict__ zeros, unsigned short* __restrict__ wbf) {
  const int gid = blockIdx.x * 256 + threadIdx.x;   // one thread per 32 elems
  const int row = gid >> 7;                         // IN_F/32 = 128 chunks/row
  const int kc  = (gid & 127) * 32;
  const int g   = kc >> 7;
  const float s  = scales[(size_t)row * NGRP + g];
  const float zs = -zeros[(size_t)row * NGRP + g] * s;
  const int* q = qweight + (size_t)row * IN_F + kc;
  uint4* o = (uint4*)(wbf + (size_t)row * IN_F + kc);
  o[0] = dq8(q, s, zs);      o[1] = dq8(q + 8, s, zs);
  o[2] = dq8(q + 16, s, zs); o[3] = dq8(q + 24, s, zs);
}

// ---------------------------------------------------------------------------
// Pre-pass C: lora_B fp32 -> bf16 (OUT_F x RANK, 128KB)
// ---------------------------------------------------------------------------
__global__ __launch_bounds__(256) void convert_b_kernel(
    const float* __restrict__ lora_B, unsigned short* __restrict__ bbf) {
  size_t i = ((size_t)blockIdx.x * 256 + threadIdx.x) * 16;
  uint4* o = (uint4*)(bbf + i);
  o[0] = pack8(lora_B + i);
  o[1] = pack8(lora_B + i + 8);
}

// ---------------------------------------------------------------------------
// t[m, r] = sum_k x[m,k] * (lora_A[r,k] * lora_E[r] * lscale)
// bf16-x variant (fast path): A frags are direct b128 loads from xbf.
// ---------------------------------------------------------------------------
__global__ __launch_bounds__(256) void adalora_t_bf16_kernel(
    const unsigned short* __restrict__ xbf, const float* __restrict__ lora_A,
    const float* __restrict__ lora_E, unsigned short* __restrict__ tbuf,
    int tokens, float lscale) {
  const int wave = threadIdx.x >> 5;
  const int lane = threadIdx.x & 31;
  const int half = lane >> 4;
  const int r    = lane & 15;
  const int m0   = (blockIdx.x * 8 + wave) * 16;
  if (m0 >= tokens) return;

  const unsigned short* xrow = xbf + (size_t)(m0 + r) * IN_F;
  const float* arow = lora_A + (size_t)r * IN_F;
  const float  es   = lora_E[r] * lscale;

  v8f acc = {};
  for (int kk = 0; kk < IN_F; kk += 32) {
    Frag a, b;
    a.u[0] = *(const uint4*)&xrow[kk + half * 8];
    a.u[1] = *(const uint4*)&xrow[kk + 16 + half * 8];
    b.u[0] = pack8s(arow + kk + half * 8, es);
    b.u[1] = pack8s(arow + kk + 16 + half * 8, es);
    acc = __builtin_amdgcn_wmma_f32_16x16x32_bf16(
        false, a.v, false, b.v, (short)0, acc, false, false);
  }
#pragma unroll
  for (int v = 0; v < 8; ++v) {
    int row = m0 + half * 8 + v;
    tbuf[(size_t)row * RANK + r] = (unsigned short)(pk2(acc[v], acc[v]) & 0xFFFFu);
  }
}

// fp32-x variant (fallback path only)
__global__ __launch_bounds__(256) void adalora_t_kernel(
    const float* __restrict__ x, const float* __restrict__ lora_A,
    const float* __restrict__ lora_E, unsigned short* __restrict__ tbuf,
    int tokens, float lscale) {
  const int wave = threadIdx.x >> 5;
  const int lane = threadIdx.x & 31;
  const int half = lane >> 4;
  const int r    = lane & 15;
  const int m0   = (blockIdx.x * 8 + wave) * 16;
  if (m0 >= tokens) return;

  const float* xrow = x + (size_t)(m0 + r) * IN_F;
  const float* arow = lora_A + (size_t)r * IN_F;
  const float  es   = lora_E[r] * lscale;

  v8f acc = {};
  for (int kk = 0; kk < IN_F; kk += 32) {
    Frag a, b;
    a.u[0] = pack8(xrow + kk + half * 8);
    a.u[1] = pack8(xrow + kk + 16 + half * 8);
    b.u[0] = pack8s(arow + kk + half * 8, es);
    b.u[1] = pack8s(arow + kk + 16 + half * 8, es);
    acc = __builtin_amdgcn_wmma_f32_16x16x32_bf16(
        false, a.v, false, b.v, (short)0, acc, false, false);
  }
#pragma unroll
  for (int v = 0; v < 8; ++v) {
    int row = m0 + half * 8 + v;
    tbuf[(size_t)row * RANK + r] = (unsigned short)(pk2(acc[v], acc[v]) & 0xFFFFu);
  }
}

// ---------------------------------------------------------------------------
// FAST GEMM: bf16 x / W / B staged by async DMA; zero conversion anywhere.
// 128x128 tile/block, 8 waves (4m x 2n), 8 acc tiles/wave, double-buffered
// LDS, K-step 64, one s_wait_asynccnt + barrier per stage.
// ---------------------------------------------------------------------------
__global__ __launch_bounds__(256) void adalora_gemm_async_kernel(
    const unsigned short* __restrict__ xbf, const unsigned short* __restrict__ wbf,
    const unsigned short* __restrict__ bbf, const unsigned short* __restrict__ tbuf,
    float* __restrict__ out) {
  __shared__ __align__(16) unsigned short xsh[2][128 * LDSW];
  __shared__ __align__(16) unsigned short wsh[2][128 * LDSW];

  const int tid  = threadIdx.x;
  const int lane = tid & 31;
  const int wave = tid >> 5;
  const int half = lane >> 4;
  const int r    = lane & 15;
  const int wvm  = wave >> 1;
  const int wvn  = wave & 1;
  const int m_base = blockIdx.y * 128;
  const int n_base = blockIdx.x * 128;

  // staging: 2 threads per row, 32 contiguous bf16 (64B = 4 async b128s) each
  const int srow = tid >> 1;
  const int skh  = (tid & 1) * 32;
  const unsigned short* xgb = xbf + (size_t)m_base * IN_F;  // uniform bases
  const unsigned short* wgb = wbf + (size_t)n_base * IN_F;
  const unsigned row_off = (unsigned)(srow * IN_F + skh) * 2u;

  auto issue_stage = [&](int buf, int kk) {
    unsigned go = row_off + (unsigned)kk * 2u;
    unsigned short* xd = &xsh[buf][srow * LDSW + skh];
    unsigned short* wd = &wsh[buf][srow * LDSW + skh];
#pragma unroll
    for (int i = 0; i < 4; ++i) {
      async_cp16(xd + i * 8, xgb, go + i * 16);
      async_cp16(wd + i * 8, wgb, go + i * 16);
    }
  };

  v8f acc[2][4];
  const v8f vzero = {};
#pragma unroll
  for (int mt = 0; mt < 2; ++mt)
#pragma unroll
    for (int nt = 0; nt < 4; ++nt) acc[mt][nt] = vzero;

  issue_stage(0, 0);
  int cur = 0;
  for (int kk = 0; kk < IN_F; kk += 64) {
    wait_async0();                 // this wave's tile landed in LDS
    __syncthreads();               // all waves' tiles landed; prev reads done
    if (kk + 64 < IN_F) issue_stage(cur ^ 1, kk + 64);

    const unsigned short* xb = xsh[cur];
    const unsigned short* wb = wsh[cur];
#pragma unroll
    for (int ks = 0; ks < 64; ks += 32) {
      Frag a[2], b[4];
#pragma unroll
      for (int mt = 0; mt < 2; ++mt) {
        int row = wvm * 32 + mt * 16 + r;
        a[mt].u[0] = *(const uint4*)&xb[row * LDSW + ks + half * 8];
        a[mt].u[1] = *(const uint4*)&xb[row * LDSW + ks + 16 + half * 8];
      }
#pragma unroll
      for (int nt = 0; nt < 4; ++nt) {
        int col = wvn * 64 + nt * 16 + r;
        b[nt].u[0] = *(const uint4*)&wb[col * LDSW + ks + half * 8];
        b[nt].u[1] = *(const uint4*)&wb[col * LDSW + ks + 16 + half * 8];
      }
#pragma unroll
      for (int mt = 0; mt < 2; ++mt)
#pragma unroll
        for (int nt = 0; nt < 4; ++nt)
          acc[mt][nt] = __builtin_amdgcn_wmma_f32_16x16x32_bf16(
              false, a[mt].v, false, b[nt].v, (short)0, acc[mt][nt], false, false);
    }
    cur ^= 1;
  }

  // LoRA epilogue: one extra WMMA per tile, K = rank(16) zero-padded to 32
  const uint4 z4 = {0u, 0u, 0u, 0u};
  Frag al[2], bl[4];
#pragma unroll
  for (int mt = 0; mt < 2; ++mt) {
    int row = m_base + wvm * 32 + mt * 16 + r;
    al[mt].u[0] = *(const uint4*)&tbuf[(size_t)row * RANK + half * 8];
    al[mt].u[1] = z4;
  }
#pragma unroll
  for (int nt = 0; nt < 4; ++nt) {
    const unsigned short* brow = bbf + (size_t)(n_base + wvn * 64 + nt * 16 + r) * RANK;
    bl[nt].u[0] = *(const uint4*)&brow[half * 8];  // elements 8-15: K=16..31 -> 0
    bl[nt].u[1] = z4;
  }
#pragma unroll
  for (int mt = 0; mt < 2; ++mt)
#pragma unroll
    for (int nt = 0; nt < 4; ++nt)
      acc[mt][nt] = __builtin_amdgcn_wmma_f32_16x16x32_bf16(
          false, al[mt].v, false, bl[nt].v, (short)0, acc[mt][nt], false, false);

#pragma unroll
  for (int mt = 0; mt < 2; ++mt)
#pragma unroll
    for (int nt = 0; nt < 4; ++nt) {
      size_t col = (size_t)(n_base + wvn * 64 + nt * 16 + r);
#pragma unroll
      for (int v = 0; v < 8; ++v) {
        size_t row = (size_t)(m_base + wvm * 32 + mt * 16 + half * 8 + v);
        out[row * OUT_F + col] = acc[mt][nt][v];
      }
    }
}

// ---------------------------------------------------------------------------
// FALLBACK fused GEMM (small ws): dequant/convert in-loop.
// ---------------------------------------------------------------------------
__global__ __launch_bounds__(256) void adalora_gemm_fused_kernel(
    const float* __restrict__ x, const int* __restrict__ qweight,
    const float* __restrict__ scales, const float* __restrict__ zeros,
    const float* __restrict__ lora_B, const unsigned short* __restrict__ tbuf,
    float* __restrict__ out) {
  __shared__ __align__(16) unsigned short xsh[2][128 * LDSW];
  __shared__ __align__(16) unsigned short wsh[2][128 * LDSW];

  const int tid  = threadIdx.x;
  const int lane = tid & 31;
  const int wave = tid >> 5;
  const int half = lane >> 4;
  const int r    = lane & 15;
  const int wvm  = wave >> 1;
  const int wvn  = wave & 1;
  const int m_base = blockIdx.y * 128;
  const int n_base = blockIdx.x * 128;

  const int srow = tid >> 1;
  const int skh  = (tid & 1) * 32;
  const float* xp0 = x + (size_t)(m_base + srow) * IN_F + skh;
  const int*   qp0 = qweight + (size_t)(n_base + srow) * IN_F + skh;
  const float* sp = scales + (size_t)(n_base + srow) * NGRP;
  const float* zp = zeros + (size_t)(n_base + srow) * NGRP;

  v8f acc[2][4];
  const v8f vzero = {};
#pragma unroll
  for (int mt = 0; mt < 2; ++mt)
#pragma unroll
    for (int nt = 0; nt < 4; ++nt) acc[mt][nt] = vzero;

  auto loadX = [&](int kk, uint4 rx[4]) {
    const float* p = xp0 + kk;
    rx[0] = pack8(p); rx[1] = pack8(p + 8);
    rx[2] = pack8(p + 16); rx[3] = pack8(p + 24);
  };
  auto loadW = [&](int kk, uint4 rw[4]) {
    const int g = kk >> 7;
    const float s = sp[g];
    const float zs = -zp[g] * s;
    const int* q = qp0 + kk;
    rw[0] = dq8(q, s, zs); rw[1] = dq8(q + 8, s, zs);
    rw[2] = dq8(q + 16, s, zs); rw[3] = dq8(q + 24, s, zs);
  };
  auto stage = [&](int buf, const uint4 rx[4], const uint4 rw[4]) {
    uint4* xd = (uint4*)&xsh[buf][srow * LDSW + skh];
    uint4* wd = (uint4*)&wsh[buf][srow * LDSW + skh];
    xd[0] = rx[0]; xd[1] = rx[1]; xd[2] = rx[2]; xd[3] = rx[3];
    wd[0] = rw[0]; wd[1] = rw[1]; wd[2] = rw[2]; wd[3] = rw[3];
  };

  {
    uint4 rx[4], rw[4];
    loadX(0, rx); loadW(0, rw);
    stage(0, rx, rw);
  }

  int cur = 0;
  for (int kk = 0; kk < IN_F; kk += 64) {
    const bool has_next = (kk + 64 < IN_F);
    uint4 nx[4], nw[4];
    if (has_next) { loadX(kk + 64, nx); loadW(kk + 64, nw); }
    __syncthreads();
    if (has_next) stage(cur ^ 1, nx, nw);

    const unsigned short* xb = xsh[cur];
    const unsigned short* wb = wsh[cur];
#pragma unroll
    for (int ks = 0; ks < 64; ks += 32) {
      Frag a[2], b[4];
#pragma unroll
      for (int mt = 0; mt < 2; ++mt) {
        int row = wvm * 32 + mt * 16 + r;
        a[mt].u[0] = *(const uint4*)&xb[row * LDSW + ks + half * 8];
        a[mt].u[1] = *(const uint4*)&xb[row * LDSW + ks + 16 + half * 8];
      }
#pragma unroll
      for (int nt = 0; nt < 4; ++nt) {
        int col = wvn * 64 + nt * 16 + r;
        b[nt].u[0] = *(const uint4*)&wb[col * LDSW + ks + half * 8];
        b[nt].u[1] = *(const uint4*)&wb[col * LDSW + ks + 16 + half * 8];
      }
#pragma unroll
      for (int mt = 0; mt < 2; ++mt)
#pragma unroll
        for (int nt = 0; nt < 4; ++nt)
          acc[mt][nt] = __builtin_amdgcn_wmma_f32_16x16x32_bf16(
              false, a[mt].v, false, b[nt].v, (short)0, acc[mt][nt], false, false);
    }
    cur ^= 1;
  }

  const uint4 z4 = {0u, 0u, 0u, 0u};
  Frag al[2], bl[4];
#pragma unroll
  for (int mt = 0; mt < 2; ++mt) {
    int row = m_base + wvm * 32 + mt * 16 + r;
    al[mt].u[0] = *(const uint4*)&tbuf[(size_t)row * RANK + half * 8];
    al[mt].u[1] = z4;
  }
#pragma unroll
  for (int nt = 0; nt < 4; ++nt) {
    int col = n_base + wvn * 64 + nt * 16 + r;
    uint4 blo = pack8(lora_B + (size_t)col * RANK + half * 8);
    bl[nt].u[0] = blo;
    bl[nt].u[1] = z4;
  }
#pragma unroll
  for (int mt = 0; mt < 2; ++mt)
#pragma unroll
    for (int nt = 0; nt < 4; ++nt)
      acc[mt][nt] = __builtin_amdgcn_wmma_f32_16x16x32_bf16(
          false, al[mt].v, false, bl[nt].v, (short)0, acc[mt][nt], false, false);

#pragma unroll
  for (int mt = 0; mt < 2; ++mt)
#pragma unroll
    for (int nt = 0; nt < 4; ++nt) {
      size_t col = (size_t)(n_base + wvn * 64 + nt * 16 + r);
#pragma unroll
      for (int v = 0; v < 8; ++v) {
        size_t row = (size_t)(m_base + wvm * 32 + mt * 16 + half * 8 + v);
        out[row * OUT_F + col] = acc[mt][nt][v];
      }
    }
}

extern "C" void kernel_launch(void* const* d_in, const int* in_sizes, int n_in,
                              void* d_out, int out_size, void* d_ws, size_t ws_size,
                              hipStream_t stream) {
  const float* x       = (const float*)d_in[0];
  const int*   qweight = (const int*)d_in[1];
  const float* scales  = (const float*)d_in[2];
  const float* zeros   = (const float*)d_in[3];
  const float* lora_A  = (const float*)d_in[4];
  const float* lora_E  = (const float*)d_in[5];
  const float* lora_B  = (const float*)d_in[6];
  float* out = (float*)d_out;

  const int tokens = in_sizes[0] / IN_F;          // 8192
  const float lscale = 16.0f / (16.0f + 1e-5f);   // SCALING / RANKNUM

  // ws layout: [tbuf 256KB][xbf 64MB][wbf 32MB][bbf 128KB]
  const size_t tbuf_bytes = (size_t)tokens * RANK * 2;
  const size_t xbf_bytes  = (size_t)tokens * IN_F * 2;
  const size_t wbf_bytes  = (size_t)OUT_F * IN_F * 2;
  const size_t bbf_bytes  = (size_t)OUT_F * RANK * 2;
  unsigned short* tbuf = (unsigned short*)d_ws;
  unsigned short* xbf  = (unsigned short*)((char*)d_ws + tbuf_bytes);
  unsigned short* wbf  = (unsigned short*)((char*)d_ws + tbuf_bytes + xbf_bytes);
  unsigned short* bbf  = (unsigned short*)((char*)d_ws + tbuf_bytes + xbf_bytes + wbf_bytes);
  const bool fast = ws_size >= tbuf_bytes + xbf_bytes + wbf_bytes + bbf_bytes;

  dim3 blk(256);
  dim3 g1((tokens + 127) / 128);
  dim3 g2(OUT_F / 128, tokens / 128);

  if (fast) {
    convert_x_kernel<<<dim3((unsigned)((size_t)tokens * IN_F / 4096)), blk, 0, stream>>>(x, xbf);
    dequant_w_kernel<<<dim3((unsigned)((size_t)OUT_F * IN_F / 32 / 256)), blk, 0, stream>>>(
        qweight, scales, zeros, wbf);
    convert_b_kernel<<<dim3((unsigned)((size_t)OUT_F * RANK / 4096)), blk, 0, stream>>>(lora_B, bbf);
    adalora_t_bf16_kernel<<<g1, blk, 0, stream>>>(xbf, lora_A, lora_E, tbuf, tokens, lscale);
    adalora_gemm_async_kernel<<<g2, blk, 0, stream>>>(xbf, wbf, bbf, tbuf, out);
  } else {
    adalora_t_kernel<<<g1, blk, 0, stream>>>(x, lora_A, lora_E, tbuf, tokens, lscale);
    adalora_gemm_fused_kernel<<<g2, blk, 0, stream>>>(x, qweight, scales, zeros,
                                                      lora_B, tbuf, out);
  }
}
